// LocalAttention_41704132444717
// MI455X (gfx1250) — matrix operand order repaired and checked by reading, not compiled, not run on hardware
//
#include <hip/hip_runtime.h>
#include <hip/hip_bf16.h>
#include <math.h>

// CDNA5 / gfx1250 fused local-attention kernel.
// D = A x B + C via v_wmma_f32_16x16x32_f16 (f16 operands, f32 accumulate).

typedef __attribute__((ext_vector_type(16))) _Float16 v16h;
typedef __attribute__((ext_vector_type(8)))  float    v8f;
typedef __attribute__((ext_vector_type(4)))  float    f32x4;

#define BB   8
#define QQ   1024
#define NN   32
#define DIMM 128
#define HH   8
#define HDIM 16
#define QB   2            // q values per workgroup
#define ROWS (QB * NN)    // 64 GEMM rows (A matrix M dim)
#define SCALE 0.25f       // HD^-0.5

union AFrag { v16h v; f32x4 q[2]; };

__global__ __launch_bounds__(256, 1)
void local_attn_fused(const float* __restrict__ aq_g,    // atom_query   [B,Q,DIM]
                      const int*   __restrict__ nbr_g,   // atom_neighbor[B,Q,N,2]
                      const float* __restrict__ geo_g,   // neighbor_geometry [B,Q,N,1]
                      const float* __restrict__ nw_g,    // neighbor_weight   [B,Q,N,DIM]
                      const float* __restrict__ mask_g,  // mask [B,Q,N]
                      const float* __restrict__ Wq, const float* __restrict__ bq,
                      const float* __restrict__ Wk, const float* __restrict__ bk,
                      const float* __restrict__ Wv, const float* __restrict__ bv,
                      const float* __restrict__ Wg, const float* __restrict__ bg,
                      const float* __restrict__ gam, const float* __restrict__ bet,
                      float* __restrict__ out_attn,      // [B,H,Q,N]
                      float* __restrict__ out_ln,        // [B,Q,DIM]
                      float* __restrict__ out_ng)        // [B,Q,N,DIM]
{
  // ---- LDS staging (~167 KB; CDNA5 allows up to 320 KB per workgroup) ----
  __shared__ __align__(16) _Float16 s_wk [DIMM * DIMM];   // Wk as f16   (32 KB)
  __shared__ __align__(16) _Float16 s_wv [DIMM * DIMM];   // Wv as f16   (32 KB)
  __shared__ __align__(16) _Float16 s_ang[ROWS * DIMM];   // A for k-GEMM (16 KB)
  __shared__ __align__(16) _Float16 s_an [ROWS * DIMM];   // A for v-GEMM (16 KB)
  __shared__ __align__(16) float    s_kbuf[ROWS * DIMM];  // k  tiles f32 (32 KB)
  __shared__ __align__(16) float    s_vbuf[ROWS * DIMM];  // vk tiles f32 (32 KB)
  __shared__ float s_aqr [QB * DIMM];
  __shared__ float s_qp  [QB * DIMM];
  __shared__ float s_attn[QB * HH * NN];
  __shared__ float s_ctx [QB * DIMM];
  __shared__ float s_bk[DIMM];
  __shared__ float s_bv[DIMM];
  __shared__ int   s_nbr [ROWS];
  __shared__ float s_geo [ROWS];
  __shared__ float s_mask[ROWS];

  const int tid = threadIdx.x;
  const int b   = blockIdx.x >> 9;          // Q/QB = 512 blocks per batch
  const int qb  = (blockIdx.x & 511) * QB;

  // ---------------- Phase 0: stage weights / biases / tile metadata ----------------
  for (int i = tid; i < DIMM * DIMM; i += 256) {
    s_wk[i] = (_Float16)Wk[i];
    s_wv[i] = (_Float16)Wv[i];
  }
  if (tid < DIMM) { s_bk[tid] = bk[tid]; s_bv[tid] = bv[tid]; }
  if (tid < ROWS) {
    int q = tid >> 5, n = tid & 31;
    int gi = (b * QQ + qb + q) * NN + n;
    s_nbr[tid]  = nbr_g[gi * 2 + 1];
    s_geo[tid]  = geo_g[gi];
    s_mask[tid] = mask_g[gi];
  }
  { int q = tid >> 7, d = tid & 127;
    s_aqr[tid] = aq_g[(b * QQ + qb + q) * DIMM + d]; }
  __syncthreads();

  // ------- Phase 1: gather + swish geometry filter; build f16 A operands; q-proj -------
  for (int i = tid; i < ROWS * DIMM; i += 256) {
    int qn = i >> 7, d = i & 127;      // qn in [0,64): (q,n) pair
    int q  = qn >> 5, n = qn & 31;
    float x  = s_geo[qn] * Wg[d] + bg[d];
    float sw = x / (1.f + __expf(-x));               // swish(x) = x*sigmoid(x)
    int gi = ((b * QQ + qb + q) * NN + n) * DIMM + d;
    float ngv = sw * nw_g[gi];
    out_ng[gi] = ngv;                                // stream ng straight to HBM
    float anv = aq_g[(b * QQ + s_nbr[qn]) * DIMM + d];
    s_an [qn * DIMM + d] = (_Float16)anv;
    s_ang[qn * DIMM + d] = (_Float16)(anv * ngv);
  }
  { // q = atom_query @ Wq + bq  (small: 2x128 outputs)
    int q = tid >> 7, d = tid & 127;
    float acc = bq[d];
    for (int k = 0; k < DIMM; ++k) acc += s_aqr[q * DIMM + k] * Wq[k * DIMM + d];
    s_qp[tid] = acc;
  }
  __syncthreads();

  // ---------------- Phase 2: WMMA GEMMs  k = ang@Wk + bk,  vk = an@Wv + bv ----------------
  {
    const int lane = tid & 31;
    const int wave = tid >> 5;        // 0..7
    const int hi   = lane >> 4;       // lane half
    const int lrow = lane & 15;
    const int g    = wave >> 2;       // 0: k-GEMM, 1: v-GEMM
    const int mt   = wave & 3;        // M tile (16 rows each, M=64)
    const _Float16* __restrict__ A  = g ? s_an  : s_ang;
    const _Float16* __restrict__ W  = g ? s_wv  : s_wk;
    const float*    __restrict__ bi = g ? s_bv  : s_bk;
    float*          __restrict__ D  = g ? s_vbuf : s_kbuf;

    v8f acc[8];
#pragma unroll
    for (int nt = 0; nt < 8; ++nt) {           // fold bias into C
      float bval = bi[nt * 16 + lrow];
#pragma unroll
      for (int v = 0; v < 8; ++v) acc[nt][v] = bval;
    }

#pragma unroll
    for (int kk = 0; kk < 4; ++kk) {           // K = 128 in 4 steps of 32
      // A fragment (ISA 7.12.2, 16-bit A 16x32): lane row = M, halfs are
      // K = kk*32 + 8*hi + [0..7] then K = kk*32 + 16 + 8*hi + [0..7]
      AFrag a;
      const _Float16* ap = &A[(mt * 16 + lrow) * DIMM + kk * 32 + 8 * hi];
      a.q[0] = *(const f32x4*)(ap);
      a.q[1] = *(const f32x4*)(ap + 16);
      // B fragment: lane column N = nt*16 + lrow, halfs are K in the same order as A
#pragma unroll
      for (int nt = 0; nt < 8; ++nt) {
        AFrag bf;
        const _Float16* bp = &W[(kk * 32 + 8 * hi) * DIMM + nt * 16 + lrow];
#pragma unroll
        for (int s = 0; s < 8; ++s) { bf.v[s] = bp[s * DIMM]; bf.v[8 + s] = bp[(16 + s) * DIMM]; }
        acc[nt] = __builtin_amdgcn_wmma_f32_16x16x32_f16(
            false, a.v, false, bf.v, (short)0, acc[nt], false, false);
      }
    }

    // D layout: lane -> N = lrow, vgpr v -> M = v + 8*hi
#pragma unroll
    for (int nt = 0; nt < 8; ++nt)
#pragma unroll
      for (int v = 0; v < 8; ++v)
        D[(mt * 16 + v + 8 * hi) * DIMM + nt * 16 + lrow] = acc[nt][v];
  }
  __syncthreads();

  // ---------------- Phase 3: per-head energies + masked softmax ----------------
  if (tid < QB * HH) {
    int q = tid >> 3, h = tid & 7;
    float e[NN];
    float mx = -3.4e38f;
    for (int n = 0; n < NN; ++n) {
      float s = 0.f;
      for (int d2 = 0; d2 < HDIM; ++d2)
        s += s_qp[q * DIMM + h * HDIM + d2] * s_kbuf[(q * NN + n) * DIMM + h * HDIM + d2];
      s = s * SCALE + (1.f - s_mask[q * NN + n]) * (-1e9f);
      e[n] = s;
      mx = fmaxf(mx, s);
    }
    float sum = 0.f;
    for (int n = 0; n < NN; ++n) { e[n] = __expf(e[n] - mx); sum += e[n]; }
    float inv = 1.f / sum;
    for (int n = 0; n < NN; ++n) {
      float a = e[n] * inv;
      s_attn[(q * HH + h) * NN + n] = a;
      out_attn[((b * HH + h) * QQ + (qb + q)) * NN + n] = a;
    }
  }
  __syncthreads();

  // ---------------- Phase 4: context = mask*attn*vk summed over n, + vq ----------------
  {
    int q = tid >> 7, d = tid & 127, h = d >> 4;
    float acc = 0.f;
    for (int n = 0; n < NN; ++n)
      acc += s_mask[q * NN + n] * s_attn[(q * HH + h) * NN + n]
           * s_vbuf[(q * NN + n) * DIMM + d];
    float vq = s_bv[d];
    for (int k = 0; k < DIMM; ++k)
      vq += s_aqr[q * DIMM + k] * (float)s_wv[k * DIMM + d];
    s_ctx[tid] = acc + vq;
  }
  __syncthreads();

  // ---------------- Phase 5: LayerNorm over DIM ----------------
  {
    int q = tid >> 7, d = tid & 127;
    float sum = 0.f;
    for (int j = 0; j < DIMM; ++j) sum += s_ctx[q * DIMM + j];
    float mu = sum * (1.f / DIMM);
    float var = 0.f;
    for (int j = 0; j < DIMM; ++j) {
      float t = s_ctx[q * DIMM + j] - mu;
      var += t * t;
    }
    var *= (1.f / DIMM);
    float inv = rsqrtf(var + 1e-6f);
    out_ln[(b * QQ + qb + q) * DIMM + d] =
        (s_ctx[q * DIMM + d] - mu) * inv * gam[d] + bet[d];
  }
}

extern "C" void kernel_launch(void* const* d_in, const int* in_sizes, int n_in,
                              void* d_out, int out_size, void* d_ws, size_t ws_size,
                              hipStream_t stream) {
  const float* aq   = (const float*)d_in[0];
  const int*   nbr  = (const int*)  d_in[1];
  const float* geo  = (const float*)d_in[2];
  const float* nw   = (const float*)d_in[3];
  const float* mask = (const float*)d_in[4];
  const float* Wq   = (const float*)d_in[5];
  const float* bq   = (const float*)d_in[6];
  const float* Wk   = (const float*)d_in[7];
  const float* bk   = (const float*)d_in[8];
  const float* Wv   = (const float*)d_in[9];
  const float* bv   = (const float*)d_in[10];
  const float* Wg   = (const float*)d_in[11];
  const float* bg   = (const float*)d_in[12];
  const float* gam  = (const float*)d_in[13];
  const float* bet  = (const float*)d_in[14];

  float* out = (float*)d_out;
  float* out_attn = out;                                   // [B,H,Q,N]  = 2 097 152
  float* out_ln   = out + (size_t)BB * HH * QQ * NN;       // [B,Q,DIM]  = 1 048 576
  float* out_ng   = out_ln + (size_t)BB * QQ * DIMM;       // [B,Q,N,DIM]= 33 554 432

  dim3 grid(BB * (QQ / QB));   // 4096 workgroups
  dim3 block(256);             // 8 wave32 per workgroup
  local_attn_fused<<<grid, block, 0, stream>>>(
      aq, nbr, geo, nw, mask, Wq, bq, Wk, bk, Wv, bv, Wg, bg, gam, bet,
      out_attn, out_ln, out_ng);
}
